// DagnabbitAutoEncoder_85349590106461
// MI455X (gfx1250) — matrix operand order repaired and checked by reading, not compiled, not run on hardware
//
#include <hip/hip_runtime.h>
#include <hip/hip_bf16.h>
#include <math.h>

// ---------------- problem constants (match reference) ----------------
#define N_NODES   16384
#define D_EMB     256
#define TWO_D     512
#define H_DIM     1024
#define NUM_ROOTS 64
#define NUM_TRUNK 32
#define NUM_ENC   33          // 32 trunk encoders + 1 shared output encoder
#define OUT_START 96          // NUM_TRUNK + NUM_ROOTS
#define MAXL      512         // level clamp (real depth ~O(ln N) ~ 40)
#define GKEYS     (NUM_ENC * MAXL)
#define PADCAP    (N_NODES + 15 * GKEYS)

// ---------------- workspace layout ----------------
#define WS_LEVEL 0
#define WS_CNT   (WS_LEVEL + N_NODES)
#define WS_BASE  (WS_CNT + GKEYS)
#define WS_ORDER (WS_BASE + GKEYS)
#define WS_LVL   (WS_ORDER + PADCAP)
#define WS_HDR   (WS_LVL + MAXL + 2)   // [0]=nlev [1]=total [2]=bar_cnt [3]=bar_gen
#define WS_TOTAL (WS_HDR + 4)
// optional f16 weight cache (pre-swizzled WMMA B-fragment layout):
#define W1C_PER_T (64 * 16 * 512)      // 64 col-tiles x 16 k-steps x (32 lanes x 16 halves)
#define W2C_PER_T (16 * 32 * 512)
#define W1C_TOTAL ((size_t)NUM_ENC * W1C_PER_T)   // 17,301,504 halves
#define W2C_TOTAL ((size_t)NUM_ENC * W2C_PER_T)   //  8,650,752 halves

#define NBLK 24
#define TPB  256

typedef __attribute__((ext_vector_type(16))) _Float16 v16h;
typedef __attribute__((ext_vector_type(8)))  _Float16 v8h;
typedef __attribute__((ext_vector_type(8)))  float    v8f;

// ============================================================
// Kernel 1: sequential level assignment + counting sort by
// (level, encoder), each group padded to a multiple of 16 so
// every 16-node tile is type-uniform for WMMA.
// ============================================================
__global__ void dag_prep_kernel(const int* __restrict__ nin,
                                const int* __restrict__ nty,
                                int* __restrict__ ws) {
  if (blockIdx.x != 0 || threadIdx.x != 0) return;
  int* level = ws + WS_LEVEL;
  int* cnt   = ws + WS_CNT;
  int* base  = ws + WS_BASE;
  int* order = ws + WS_ORDER;
  int* lvb   = ws + WS_LVL;
  int* hdr   = ws + WS_HDR;
  hdr[2] = 0; hdr[3] = 0;                       // reset grid barrier
  for (int i = 0; i < GKEYS; ++i) cnt[i] = 0;
  for (int n = 0; n < NUM_ROOTS; ++n) level[n] = 0;
  int nlev = 0;
  for (int n = NUM_ROOTS; n < N_NODES; ++n) {
    int ty = nty[n];
    int i0 = nin[2 * n];
    int lv;
    if (ty >= OUT_START) {
      lv = level[i0] + 1;                       // output node: single parent
    } else {
      int l1 = level[nin[2 * n + 1]];
      int l0 = level[i0];
      lv = (l0 > l1 ? l0 : l1) + 1;
    }
    if (lv > MAXL - 1) lv = MAXL - 1;
    level[n] = lv;
    if (lv > nlev) nlev = lv;
    int pidx = (ty >= OUT_START) ? NUM_TRUNK : (ty < 0 ? 0 : (ty > 31 ? 31 : ty));
    cnt[lv * NUM_ENC + pidx]++;
  }
  int run = 0;
  for (int l = 0; l <= nlev; ++l) {
    lvb[l] = run;
    for (int t = 0; t < NUM_ENC; ++t) {
      int k = l * NUM_ENC + t;
      base[k] = run;
      run += (cnt[k] + 15) & ~15;               // pad each group to 16
    }
  }
  lvb[nlev + 1] = run;
  hdr[0] = nlev; hdr[1] = run;
  for (int i = 0; i < run; ++i) order[i] = -1;  // padding sentinel
  for (int i = 0; i < GKEYS; ++i) cnt[i] = base[i];
  for (int n = NUM_ROOTS; n < N_NODES; ++n) {
    int ty = nty[n];
    int pidx = (ty >= OUT_START) ? NUM_TRUNK : ty;
    int k = level[n] * NUM_ENC + pidx;
    order[cnt[k]++] = n;
  }
}

// ============================================================
// Kernel 1b (optional): one-time f32 -> f16 weight swizzle into
// the per-lane WMMA B-fragment layout (lane's 16 halves = 32B,
// so a B fragment is two global_load_b128).
// ============================================================
__global__ void weight_swizzle_kernel(const float* __restrict__ W1,
                                      const float* __restrict__ W2,
                                      _Float16* __restrict__ W1c,
                                      _Float16* __restrict__ W2c) {
  const size_t stride = (size_t)gridDim.x * blockDim.x;
  for (size_t i = (size_t)blockIdx.x * blockDim.x + threadIdx.x; i < W1C_TOTAL; i += stride) {
    const int e = (int)(i & 15), lane = (int)((i >> 4) & 31);
    const int ks = (int)((i >> 9) & 15), ct = (int)((i >> 13) & 63), t = (int)(i >> 19);
    const size_t k   = (size_t)ks * 32 + (lane >> 4) * 16 + e;
    const size_t col = (size_t)ct * 16 + (lane & 15);
    W1c[i] = (_Float16)W1[((size_t)t * TWO_D + k) * H_DIM + col];
  }
  for (size_t i = (size_t)blockIdx.x * blockDim.x + threadIdx.x; i < W2C_TOTAL; i += stride) {
    const int e = (int)(i & 15), lane = (int)((i >> 4) & 31);
    const int ks = (int)((i >> 9) & 31), ct = (int)((i >> 14) & 15), t = (int)(i >> 18);
    const size_t k   = (size_t)ks * 32 + (lane >> 4) * 16 + e;
    const size_t col = (size_t)ct * 16 + (lane & 15);
    W2c[i] = (_Float16)W2[((size_t)t * H_DIM + k) * D_EMB + col];
  }
}

// ---------------- device-scope grid barrier ----------------
__device__ __forceinline__ void grid_barrier(int* bcnt, int* bgen, int& mygen) {
  __syncthreads();
  __threadfence();                                  // release buf stores
  if (threadIdx.x == 0) {
    if (atomicAdd(bcnt, 1) == NBLK - 1) {
      __hip_atomic_store(bcnt, 0, __ATOMIC_RELAXED, __HIP_MEMORY_SCOPE_AGENT);
      __hip_atomic_fetch_add(bgen, 1, __ATOMIC_RELEASE, __HIP_MEMORY_SCOPE_AGENT);
    } else {
      while (__hip_atomic_load(bgen, __ATOMIC_ACQUIRE, __HIP_MEMORY_SCOPE_AGENT) <= mygen)
        __builtin_amdgcn_s_sleep(2);
    }
  }
  mygen += 1;
  __syncthreads();
  __threadfence();                                  // acquire other blocks' stores
}

// ============================================================
// Kernel 2: level-synchronous sweep. One block processes 16-node
// type-uniform tiles: X[16x512](f16,LDS) @ W1 -> GELU -> H[16x1024]
// (f16,LDS) @ W2 -> +b2 -> normalize*16 -> buf, all via
// v_wmma_f32_16x16x32_f16. CACHED path software-pipelines the
// pre-swizzled B-fragment loads (lookahead 2) so WMMAs never wait
// on loadcnt==0.
// ============================================================
template <bool CACHED>
__global__ __launch_bounds__(TPB) void dag_sweep_kernel(
    const int* __restrict__ nin, const int* __restrict__ nty,
    const float* __restrict__ root_emb, const float* __restrict__ slot_emb,
    const float* __restrict__ W1, const float* __restrict__ b1,
    const float* __restrict__ W2, const float* __restrict__ b2,
    const _Float16* __restrict__ W1c, const _Float16* __restrict__ W2c,
    float* __restrict__ buf, int* __restrict__ ws)
{
  __shared__ __align__(16) unsigned char s_xy[16 * TWO_D * 2];  // 16KB: X(f16) aliased with Y(f32)
  __shared__ __align__(16) _Float16 s_h[16 * H_DIM];            // 32KB
  __shared__ int   s_node[16], s_p0[16], s_p1[16], s_pidx;
  __shared__ float s_norm[16];

  _Float16* sX = (_Float16*)s_xy;
  float*    sY = (float*)s_xy;

  const int tid  = threadIdx.x;
  const int lane = tid & 31;
  const int wv   = tid >> 5;      // wave 0..7
  const int Ml   = lane & 15;     // matrix row / column-within-tile
  const int hi   = lane >> 4;     // lane half select

  int* order = ws + WS_ORDER;
  int* lvb   = ws + WS_LVL;
  int* hdr   = ws + WS_HDR;
  int* bcnt  = hdr + 2;
  int* bgen  = hdr + 3;
  int  mygen = 0;

  // -------- phase 0: seed root embeddings into buf --------
  for (int i = blockIdx.x * TPB + tid; i < NUM_ROOTS * D_EMB; i += NBLK * TPB)
    buf[i] = root_emb[i];
  grid_barrier(bcnt, bgen, mygen);

  const int nlev = hdr[0];

  for (int l = 1; l <= nlev; ++l) {
    const int beg    = lvb[l];
    const int ntiles = (lvb[l + 1] - beg) >> 4;
    for (int t = blockIdx.x; t < ntiles; t += NBLK) {
      const int tb = beg + t * 16;

      // -------- gather tile metadata + X[16x512] into LDS (f16) --------
      if (tid < 16) {
        int node = order[tb + tid];
        s_node[tid] = node;
        int p0 = -1, p1 = -1;
        if (node >= 0) {
          int ty = nty[node];
          p0 = nin[2 * node];
          p1 = (ty >= OUT_START) ? (-2 - (ty - OUT_START)) : nin[2 * node + 1];
        }
        s_p0[tid] = p0; s_p1[tid] = p1;
        if (tid == 0) {
          int ty0 = nty[node];   // row 0 of a tile is always a real node
          s_pidx = (ty0 >= OUT_START) ? NUM_TRUNK : ty0;
        }
      }
      __syncthreads();
      {
        const int row = tid >> 4, chunk = tid & 15;
        const int node = s_node[row];
        _Float16* dst = sX + row * TWO_D + chunk * 32;
        if (node < 0) {
          #pragma unroll
          for (int j = 0; j < 32; ++j) dst[j] = (_Float16)0.0f;
        } else {
          int c = chunk * 32;
          const float* src;
          if (c < D_EMB) {
            src = buf + (size_t)s_p0[row] * D_EMB + c;
          } else {
            int p1 = s_p1[row];
            c -= D_EMB;
            src = (p1 >= 0) ? (buf + (size_t)p1 * D_EMB + c)
                            : (slot_emb + (size_t)(-2 - p1) * D_EMB + c);
          }
          #pragma unroll
          for (int j = 0; j < 32; ++j) dst[j] = (_Float16)src[j];
        }
      }
      __syncthreads();

      const int pidx = s_pidx;

      // -------- layer 1: H = GELU(X @ W1 + b1), wave owns 128 cols --------
      {
        const float*    W1t  = W1 + (size_t)pidx * TWO_D * H_DIM;
        const _Float16* W1ct = W1c + (size_t)pidx * W1C_PER_T;
        v8f acc[8];
        #pragma unroll
        for (int q = 0; q < 8; ++q)
          #pragma unroll
          for (int r = 0; r < 8; ++r) acc[q][r] = 0.0f;

        auto ldA1 = [&](int kk) -> v16h {
          const _Float16* ar = sX + Ml * TWO_D + kk * 32 + hi * 16;
          v8h alo = *(const v8h*)ar;
          v8h ahi = *(const v8h*)(ar + 8);
          v16h a;
          #pragma unroll
          for (int e = 0; e < 8; ++e) { a[e] = alo[e]; a[8 + e] = ahi[e]; }
          return a;
        };

        if constexpr (CACHED) {
          auto ldB1 = [&](int q, int kk) -> v16h {
            return *(const v16h*)(W1ct + ((((size_t)(wv * 8 + q)) * 16 + kk) * 512) + lane * 16);
          };
          v16h a_cur = ldA1(0);
          v16h b_cur = ldB1(0, 0);
          v16h b_nxt = ldB1(1, 0);
          #pragma unroll 1
          for (int ks = 0; ks < TWO_D / 32; ++ks) {
            const int ksn = (ks + 1 < TWO_D / 32) ? ks + 1 : ks;  // clamped lookahead
            v16h a_nxt = ldA1(ksn);
            #pragma unroll
            for (int q = 0; q < 8; ++q) {
              const int qa = (q + 2) & 7;
              v16h b_pre = ldB1(qa, (q + 2 >= 8) ? ksn : ks);     // lookahead 2 fragments
              acc[q] = __builtin_amdgcn_wmma_f32_16x16x32_f16(
                  false, a_cur, false, b_cur, (short)0, acc[q], false, false);
              b_cur = b_nxt; b_nxt = b_pre;
            }
            a_cur = a_nxt;
          }
        } else {
          #pragma unroll 1
          for (int ks = 0; ks < TWO_D / 32; ++ks) {
            v16h a = ldA1(ks);
            const int klo = ks * 32 + hi * 16;
            if (ks + 1 < TWO_D / 32)
              __builtin_prefetch(W1t + (size_t)(klo + 32) * H_DIM + wv * 128 + Ml, 0, 0);
            #pragma unroll
            for (int q = 0; q < 8; ++q) {
              const int col = (wv * 8 + q) * 16 + Ml;
              const float* wp = W1t + (size_t)klo * H_DIM + col;
              v16h b;
              #pragma unroll
              for (int e = 0; e < 16; ++e) b[e] = (_Float16)wp[(size_t)e * H_DIM];
              acc[q] = __builtin_amdgcn_wmma_f32_16x16x32_f16(
                  false, a, false, b, (short)0, acc[q], false, false);
            }
          }
        }
        #pragma unroll
        for (int q = 0; q < 8; ++q) {
          const int col = (wv * 8 + q) * 16 + Ml;
          const float bb = b1[pidx * H_DIM + col];
          #pragma unroll
          for (int r = 0; r < 8; ++r) {
            const float h = acc[q][r] + bb;
            const float g = 0.5f * h * (1.0f + erff(h * 0.70710678f));  // exact GELU
            s_h[(r + hi * 8) * H_DIM + col] = (_Float16)g;
          }
        }
      }
      __syncthreads();

      // -------- layer 2: Y = H @ W2 + b2, wave owns 32 cols --------
      {
        const float*    W2t  = W2 + (size_t)pidx * H_DIM * D_EMB;
        const _Float16* W2ct = W2c + (size_t)pidx * W2C_PER_T;
        v8f acc2[2];
        #pragma unroll
        for (int q = 0; q < 2; ++q)
          #pragma unroll
          for (int r = 0; r < 8; ++r) acc2[q][r] = 0.0f;

        auto ldA2 = [&](int kk) -> v16h {
          const _Float16* ar = s_h + Ml * H_DIM + kk * 32 + hi * 16;
          v8h alo = *(const v8h*)ar;
          v8h ahi = *(const v8h*)(ar + 8);
          v16h a;
          #pragma unroll
          for (int e = 0; e < 8; ++e) { a[e] = alo[e]; a[8 + e] = ahi[e]; }
          return a;
        };

        if constexpr (CACHED) {
          auto ldB2 = [&](int q, int kk) -> v16h {
            return *(const v16h*)(W2ct + ((((size_t)(wv * 2 + q)) * 32 + kk) * 512) + lane * 16);
          };
          v16h a_cur = ldA2(0);
          v16h b_cur = ldB2(0, 0);
          v16h b_nxt = ldB2(1, 0);
          #pragma unroll 1
          for (int ks = 0; ks < H_DIM / 32; ++ks) {
            const int ksn = (ks + 1 < H_DIM / 32) ? ks + 1 : ks;
            v16h a_nxt = ldA2(ksn);
            #pragma unroll
            for (int q = 0; q < 2; ++q) {
              v16h b_pre = ldB2(q, ksn);                          // lookahead 2 fragments
              acc2[q] = __builtin_amdgcn_wmma_f32_16x16x32_f16(
                  false, a_cur, false, b_cur, (short)0, acc2[q], false, false);
              b_cur = b_nxt; b_nxt = b_pre;
            }
            a_cur = a_nxt;
          }
        } else {
          #pragma unroll 1
          for (int ks = 0; ks < H_DIM / 32; ++ks) {
            v16h a = ldA2(ks);
            const int klo = ks * 32 + hi * 16;
            if (ks + 1 < H_DIM / 32)
              __builtin_prefetch(W2t + (size_t)(klo + 32) * D_EMB + wv * 32 + Ml, 0, 0);
            #pragma unroll
            for (int q = 0; q < 2; ++q) {
              const int col = (wv * 2 + q) * 16 + Ml;
              const float* wp = W2t + (size_t)klo * D_EMB + col;
              v16h b;
              #pragma unroll
              for (int e = 0; e < 16; ++e) b[e] = (_Float16)wp[(size_t)e * D_EMB];
              acc2[q] = __builtin_amdgcn_wmma_f32_16x16x32_f16(
                  false, a, false, b, (short)0, acc2[q], false, false);
            }
          }
        }
        #pragma unroll
        for (int q = 0; q < 2; ++q) {
          const int col = (wv * 2 + q) * 16 + Ml;
          const float bb = b2[pidx * D_EMB + col];
          #pragma unroll
          for (int r = 0; r < 8; ++r)
            sY[(r + hi * 8) * D_EMB + col] = acc2[q][r] + bb;   // sY aliases dead sX
        }
      }
      __syncthreads();

      // -------- row norms via LDS float atomics, scale, scatter --------
      if (tid < 16) s_norm[tid] = 0.0f;
      __syncthreads();
      {
        const int row = tid >> 4, seg = tid & 15;
        float p = 0.0f;
        #pragma unroll
        for (int c = 0; c < 16; ++c) {
          const float v = sY[row * D_EMB + seg * 16 + c];
          p += v * v;
        }
        atomicAdd(&s_norm[row], p);    // ds_add_f32
      }
      __syncthreads();
      {
        const int row = tid >> 4, seg = tid & 15;
        const int node = s_node[row];
        if (node >= 0) {
          const float scale = 16.0f / fmaxf(sqrtf(s_norm[row]), 1e-12f);
          float* dstp = buf + (size_t)node * D_EMB + seg * 16;
          const float* srcp = sY + row * D_EMB + seg * 16;
          #pragma unroll
          for (int c = 0; c < 16; ++c) dstp[c] = srcp[c] * scale;
        }
      }
      __syncthreads();
    }
    grid_barrier(bcnt, bgen, mygen);   // level complete
  }
}

extern "C" void kernel_launch(void* const* d_in, const int* in_sizes, int n_in,
                              void* d_out, int out_size, void* d_ws, size_t ws_size,
                              hipStream_t stream) {
  (void)in_sizes; (void)n_in; (void)out_size;
  const int*   nin  = (const int*)d_in[0];
  const int*   nty  = (const int*)d_in[1];
  const float* root = (const float*)d_in[2];
  const float* slot = (const float*)d_in[3];
  const float* W1   = (const float*)d_in[4];
  const float* b1   = (const float*)d_in[5];
  const float* W2   = (const float*)d_in[6];
  const float* b2   = (const float*)d_in[7];
  float* buf = (float*)d_out;
  int*   ws  = (int*)d_ws;

  const size_t base_bytes = (size_t)WS_TOTAL * sizeof(int);          // ~1.3 MB
  if (ws_size < base_bytes) return;
  const size_t w1c_off    = (base_bytes + 255) & ~(size_t)255;
  const size_t need_cache = w1c_off + (W1C_TOTAL + W2C_TOTAL) * sizeof(_Float16);  // ~50.7 MiB
  const bool cached = (ws_size >= need_cache);

  _Float16* W1cp = (_Float16*)((char*)d_ws + w1c_off);
  _Float16* W2cp = W1cp + W1C_TOTAL;

  hipLaunchKernelGGL(dag_prep_kernel, dim3(1), dim3(1), 0, stream, nin, nty, ws);
  if (cached) {
    hipLaunchKernelGGL(weight_swizzle_kernel, dim3(256), dim3(256), 0, stream,
                       W1, W2, W1cp, W2cp);
    hipLaunchKernelGGL(dag_sweep_kernel<true>, dim3(NBLK), dim3(TPB), 0, stream,
                       nin, nty, root, slot, W1, b1, W2, b2, W1cp, W2cp, buf, ws);
  } else {
    hipLaunchKernelGGL(dag_sweep_kernel<false>, dim3(NBLK), dim3(TPB), 0, stream,
                       nin, nty, root, slot, W1, b1, W2, b2, W1cp, W2cp, buf, ws);
  }
}
